// ModFusion_7310034338274
// MI455X (gfx1250) — matrix operand order repaired and checked by reading, not compiled, not run on hardware
//
#include <hip/hip_runtime.h>
#include <hip/hip_bf16.h>

// Problem constants (from setup_inputs)
#define BATCH 4
#define NNODE 25000
#define BN    100000          // BATCH*NNODE
#define BNPAD 100096          // padded to 6256 row tiles (782 blocks * 8 waves * 16)
#define C     128
#define RREL  8
#define KTOT  1152            // (RREL+1)*C fused GEMM K
#define NKC   36              // KTOT/32 k-chunks
#define NTPAD 6256            // BNPAD/16 row tiles
#define SOUT  100

typedef __attribute__((ext_vector_type(16))) __bf16 v16bf;
typedef __attribute__((ext_vector_type(8)))  __bf16 v8bf;
typedef __attribute__((ext_vector_type(4)))  __bf16 v4bf;
typedef __attribute__((ext_vector_type(8)))  float  v8f;

#if __has_builtin(__builtin_amdgcn_sched_barrier)
#define SCHED_FENCE() __builtin_amdgcn_sched_barrier(0)
#else
#define SCHED_FENCE()
#endif

// ---------------- LayerNorm1 -> bf16 activation, wave-per-row -----------------
__global__ __launch_bounds__(256) void k_ln1(const float* __restrict__ x,
                                             const float* __restrict__ w,
                                             const float* __restrict__ b,
                                             __bf16* __restrict__ Act) {
  const int lane = threadIdx.x & 31;
  const int row  = (blockIdx.x * 256 + threadIdx.x) >> 5;   // 8 rows per block
  float4 v = ((const float4*)x)[(size_t)row * 32 + lane];
  float s = v.x + v.y + v.z + v.w;
#pragma unroll
  for (int o = 16; o > 0; o >>= 1) s += __shfl_xor(s, o, 32);
  float mu = s * (1.0f / 128.0f);
  float dx = v.x - mu, dy = v.y - mu, dz = v.z - mu, dw = v.w - mu;
  float q = dx * dx + dy * dy + dz * dz + dw * dw;
#pragma unroll
  for (int o = 16; o > 0; o >>= 1) q += __shfl_xor(q, o, 32);
  float inv = rsqrtf(q * (1.0f / 128.0f) + 1e-5f);
  float4 w4 = ((const float4*)w)[lane];
  float4 b4 = ((const float4*)b)[lane];
  v4bf o4;
  o4[0] = (__bf16)(dx * inv * w4.x + b4.x);
  o4[1] = (__bf16)(dy * inv * w4.y + b4.y);
  o4[2] = (__bf16)(dz * inv * w4.z + b4.z);
  o4[3] = (__bf16)(dw * inv * w4.w + b4.w);
  *(v4bf*)(Act + (size_t)row * KTOT + lane * 4) = o4;
}

// ---------------- Pack [W_root ; W_rel] into bf16 [KTOT,128] ------------------
__global__ void k_wconv(const float* __restrict__ W_root,
                        const float* __restrict__ W_rel,
                        __bf16* __restrict__ Wc) {
  int gid = blockIdx.x * blockDim.x + threadIdx.x;
  if (gid >= KTOT * C) return;
  int k = gid / C, d = gid % C;
  float v = (k < C) ? W_root[k * C + d] : W_rel[(size_t)(k - C) * C + d];
  Wc[gid] = (__bf16)v;
}

// ---------------- Edge scatter: A[(dst,rel),:] += h[src,:]; cnt++ -------------
__global__ void k_scatter(const int* __restrict__ ei, const int* __restrict__ et,
                          const __bf16* __restrict__ Act,
                          float* __restrict__ A, float* __restrict__ cnt, int E) {
  long gid = (long)blockIdx.x * blockDim.x + threadIdx.x;
  long e = gid >> 5;
  if (e >= E) return;
  int q = (int)(gid & 31);
  int src = ei[e];
  int dst = ei[(long)E + e];
  int rel = et[e];
  size_t seg = (size_t)dst * RREL + rel;
  const __bf16* hp = Act + (size_t)src * KTOT + q * 4;
  float* ap = A + seg * C + q * 4;
#pragma unroll
  for (int j = 0; j < 4; ++j) atomicAdd(ap + j, (float)hp[j]);
  if (q == 0) atomicAdd(cnt + seg, 1.0f);
}

// ---------------- mean = A / max(cnt,1) -> bf16 into Act cols 128.. -----------
__global__ void k_mean(const float* __restrict__ A, const float* __restrict__ cnt,
                       __bf16* __restrict__ Act) {
  long gid = (long)blockIdx.x * blockDim.x + threadIdx.x;
  long seg = gid >> 5;
  if (seg >= (long)BN * RREL) return;
  int q = (int)(gid & 31);
  int n = (int)(seg >> 3);
  int r = (int)(seg & 7);
  float cv = cnt[seg];
  float inv = 1.0f / (cv > 1.0f ? cv : 1.0f);
  const float* ap = A + (size_t)seg * C + q * 4;
  __bf16* op = Act + (size_t)n * KTOT + C + r * C + q * 4;
#pragma unroll
  for (int j = 0; j < 4; ++j) op[j] = (__bf16)(ap[j] * inv);
}

// ---------------- Fused-K WMMA GEMM, 4-deep async-LDS ring, 1 barrier/chunk ---
// out[BNPAD,128] = Act[BNPAD,1152] @ Wc[1152,128] + bias
__global__ __launch_bounds__(256) void k_gemm(const __bf16* __restrict__ Act,
                                              const __bf16* __restrict__ Wc,
                                              const float* __restrict__ bias,
                                              float* __restrict__ out) {
  __shared__ __align__(32) __bf16 ldsB[4 * 32 * 128];   // 4 x 8KB ring of W chunks
  const int tid  = threadIdx.x;
  const int wave = tid >> 5;
  const int lane = tid & 31;
  const int wt   = blockIdx.x * 8 + wave;               // always valid (padded)
  const int rowbase = wt * 16;
  const int arow = rowbase + (lane & 15);
  // 16-bit A layout: lanes 0-15 hold K {0..7,16..23}, lanes 16-31 K {8..15,24..31}
  const size_t abase = (size_t)arow * KTOT + ((lane & 16) ? 8 : 0);

  // Low 32 bits of a shared-aperture flat pointer = workgroup-relative LDS offset
  const unsigned ldsbase = (unsigned)(uintptr_t)(void*)&ldsB[0];
  const unsigned long long gWc = (unsigned long long)(uintptr_t)Wc;

#define ISSUE_STAGE(kcs)                                                               \
  do {                                                                                 \
    unsigned long long g_ = gWc + (unsigned long long)(kcs) * 8192u + tid * 16u;       \
    unsigned l_ = ldsbase + ((unsigned)(kcs) & 3u) * 8192u + tid * 16u;                \
    asm volatile("global_load_async_to_lds_b128 %0, %1, off" :: "v"(l_), "v"(g_) : "memory"); \
    asm volatile("global_load_async_to_lds_b128 %0, %1, off" :: "v"(l_ + 4096u), "v"(g_ + 4096ull) : "memory"); \
  } while (0)

  // prologue: stages 0 and 1 in flight
  ISSUE_STAGE(0);
  ISSUE_STAGE(1);

  v8f acc[8] = {};

  for (int kc = 0; kc < NKC; ++kc) {
    // Ring safety: writes target buf[(kc+2)&3]; concurrent readers (waves at most
    // one iteration behind, enforced by the single per-iteration barrier) touch
    // buf[(kc-1)&3] or buf[kc&3] -- distinct mod 4.
    if (kc < NKC - 2) {
      ISSUE_STAGE(kc + 2);
      asm volatile("s_wait_asynccnt 4" ::: "memory");   // stage kc resident (in-order)
    } else if (kc == NKC - 2) {
      asm volatile("s_wait_asynccnt 2" ::: "memory");
    } else {
      asm volatile("s_wait_asynccnt 0" ::: "memory");
    }
    __syncthreads();   // stage kc visible to all waves of the block

    if (kc + 1 < NKC)
      __builtin_prefetch(Act + abase + (size_t)(kc + 1) * 32, 0, 3);
    v8bf a0 = *(const v8bf*)(Act + abase + (size_t)kc * 32);
    v8bf a1 = *(const v8bf*)(Act + abase + (size_t)kc * 32 + 16);
    v16bf a;
#pragma unroll
    for (int i = 0; i < 8; ++i) { a[i] = a0[i]; a[8 + i] = a1[i]; }

    // B layout: lane%16 -> N column; lanes 0-15 hold K {0..7,16..23}, lanes 16-31 K {8..15,24..31}.
    // Register-double-buffered B fragments + sched fences: keeps only 2 fragments
    // live so accumulators stay pinned (no per-iteration v_mov shuffles).
    const __bf16* bcol = ldsB + ((unsigned)kc & 3u) * 4096 + (lane & 15) + ((lane & 16) ? 8 * 128 : 0);
    v16bf bcur;
#pragma unroll
    for (int i = 0; i < 8; ++i) { bcur[i] = bcol[i * 128]; bcur[8 + i] = bcol[(16 + i) * 128]; }
#pragma unroll
    for (int nt = 0; nt < 8; ++nt) {
      v16bf bnext = bcur;
      if (nt + 1 < 8) {
#pragma unroll
        for (int i = 0; i < 8; ++i) { bnext[i] = bcol[(nt + 1) * 16 + i * 128]; bnext[8 + i] = bcol[(nt + 1) * 16 + (16 + i) * 128]; }
      }
      acc[nt] = __builtin_amdgcn_wmma_f32_16x16x32_bf16(
          false, a, false, bcur, (short)0, acc[nt], false, false);
      SCHED_FENCE();
      bcur = bnext;
    }
  }

  // D layout: VGPR r -> M = r + 8*(lane>=16); lane%16 -> N
  const int mhalf = (lane >> 4) * 8;
  const int ncol  = lane & 15;
#pragma unroll
  for (int nt = 0; nt < 8; ++nt) {
    float bv = bias[nt * 16 + ncol];
#pragma unroll
    for (int r = 0; r < 8; ++r) {
      int m = rowbase + mhalf + r;
      out[(size_t)m * C + nt * 16 + ncol] = acc[nt][r] + bv;
    }
  }
#undef ISSUE_STAGE
}

// ---------------- LN2 + relu + mean-pool over nodes, wave-per-row -------------
__global__ __launch_bounds__(256) void k_pool(const float* __restrict__ hbuf,
                                              const float* __restrict__ w,
                                              const float* __restrict__ b,
                                              float* __restrict__ feats) {
  const int lane = threadIdx.x & 31;
  const int wave = threadIdx.x >> 5;
  const int bb = blockIdx.y;
  const int rbase = blockIdx.x * 256 + wave * 32;   // 32 rows per wave
  float4 w4 = ((const float4*)w)[lane];
  float4 b4 = ((const float4*)b)[lane];
  float ax = 0.f, ay = 0.f, az = 0.f, aw = 0.f;
  for (int j = 0; j < 32; ++j) {
    int r = rbase + j;
    if (r >= NNODE) break;
    size_t row = (size_t)bb * NNODE + r;
    float4 v = ((const float4*)hbuf)[row * 32 + lane];
    float s = v.x + v.y + v.z + v.w;
#pragma unroll
    for (int o = 16; o > 0; o >>= 1) s += __shfl_xor(s, o, 32);
    float mu = s * (1.0f / 128.0f);
    float dx = v.x - mu, dy = v.y - mu, dz = v.z - mu, dw = v.w - mu;
    float q = dx * dx + dy * dy + dz * dz + dw * dw;
#pragma unroll
    for (int o = 16; o > 0; o >>= 1) q += __shfl_xor(q, o, 32);
    float inv = rsqrtf(q * (1.0f / 128.0f) + 1e-5f);
    float yx = dx * inv * w4.x + b4.x;
    float yy = dy * inv * w4.y + b4.y;
    float yz = dz * inv * w4.z + b4.z;
    float yw = dw * inv * w4.w + b4.w;
    ax += yx > 0.f ? yx : 0.f;
    ay += yy > 0.f ? yy : 0.f;
    az += yz > 0.f ? yz : 0.f;
    aw += yw > 0.f ? yw : 0.f;
  }
  const float sc = 1.0f / (float)NNODE;
  atomicAdd(&feats[bb * C + lane * 4 + 0], ax * sc);
  atomicAdd(&feats[bb * C + lane * 4 + 1], ay * sc);
  atomicAdd(&feats[bb * C + lane * 4 + 2], az * sc);
  atomicAdd(&feats[bb * C + lane * 4 + 3], aw * sc);
}

// ---------------- tiny head: [4,128]@[128,100]+b ------------------------------
__global__ void k_head(const float* __restrict__ feats, const float* __restrict__ hw,
                       const float* __restrict__ hb, float* __restrict__ out) {
  int gid = blockIdx.x * blockDim.x + threadIdx.x;
  if (gid >= BATCH * SOUT) return;
  int bb = gid / SOUT, s = gid % SOUT;
  float acc = hb[s];
#pragma unroll 8
  for (int c = 0; c < C; ++c) acc += feats[bb * C + c] * hw[c * SOUT + s];
  out[gid] = acc;
}

extern "C" void kernel_launch(void* const* d_in, const int* in_sizes, int n_in,
                              void* d_out, int out_size, void* d_ws, size_t ws_size,
                              hipStream_t stream) {
  const float* x      = (const float*)d_in[0];
  const int*   ei     = (const int*)d_in[1];
  const int*   et     = (const int*)d_in[2];
  const float* ln1_w  = (const float*)d_in[3];
  const float* ln1_b  = (const float*)d_in[4];
  const float* W_rel  = (const float*)d_in[5];
  const float* W_root = (const float*)d_in[6];
  const float* rgcn_b = (const float*)d_in[7];
  const float* ln2_w  = (const float*)d_in[8];
  const float* ln2_b  = (const float*)d_in[9];
  const float* head_w = (const float*)d_in[10];
  const float* head_b = (const float*)d_in[11];
  const int E = in_sizes[2];

  // Workspace layout (256B-aligned sections)
  char* ws = (char*)d_ws;
  size_t offAct = 0;                                    // BNPAD*KTOT bf16
  size_t offA   = offAct + (size_t)BNPAD * KTOT * 2;    // BN*R*C f32 (also reused as out)
  size_t offCnt = offA   + (size_t)BN * RREL * C * 4;
  size_t offWc  = offCnt + (size_t)BN * RREL * 4;
  size_t offFe  = offWc  + (size_t)KTOT * C * 2;

  __bf16* Act   = (__bf16*)(ws + offAct);
  float*  A     = (float*)(ws + offA);
  float*  cnt   = (float*)(ws + offCnt);
  __bf16* Wc    = (__bf16*)(ws + offWc);
  float*  feats = (float*)(ws + offFe);
  float*  hbuf  = A;   // reuse A region for RGCN output after k_mean consumed A

  hipMemsetAsync(A,     0, (size_t)BN * RREL * C * 4, stream);
  hipMemsetAsync(cnt,   0, (size_t)BN * RREL * 4,     stream);
  hipMemsetAsync(feats, 0, (size_t)BATCH * C * 4,     stream);
  // zero the 96 pad rows of Act so padded GEMM tiles are well-defined
  hipMemsetAsync(Act + (size_t)BN * KTOT, 0, (size_t)(BNPAD - BN) * KTOT * 2, stream);

  k_wconv<<<(KTOT * C + 255) / 256, 256, 0, stream>>>(W_root, W_rel, Wc);
  k_ln1<<<BN / 8, 256, 0, stream>>>(x, ln1_w, ln1_b, Act);
  {
    long tot = (long)E * 32;
    k_scatter<<<(unsigned)((tot + 255) / 256), 256, 0, stream>>>(ei, et, Act, A, cnt, E);
  }
  {
    long tot = (long)BN * RREL * 32;
    k_mean<<<(unsigned)((tot + 255) / 256), 256, 0, stream>>>(A, cnt, Act);
  }
  k_gemm<<<NTPAD / 8, 256, 0, stream>>>(Act, Wc, rgcn_b, hbuf);
  k_pool<<<dim3((NNODE + 255) / 256, BATCH), 256, 0, stream>>>(hbuf, ln2_w, ln2_b, feats);
  k_head<<<(BATCH * SOUT + 255) / 256, 256, 0, stream>>>(feats, head_w, head_b, (float*)d_out);
}
